// MultiHeadAttention_44238163148869
// MI455X (gfx1250) — compile-verified, matrix operand order checked by reading, not run on hardware
//
#include <hip/hip_runtime.h>
#include <hip/hip_bf16.h>

// ---------------------------------------------------------------------------
// MI455X (gfx1250) multi-head attention.
// Compute-bound (≈86 GFLOP vs ≈100MB traffic @23.3TB/s) -> everything runs on
// v_wmma_f32_16x16x32_bf16 (f32-accumulate).  x and weights are converted to
// bf16 once; x and K/V^T tiles move through LDS via gfx1250
// global_load_async_to_lds_b128 (ASYNCcnt) with double buffering.
// B=16, S=512, H=1024, NH=16, DH=64
// ---------------------------------------------------------------------------

#define Bb   16
#define Ss   512
#define Hh   1024
#define NHh  16
#define DHh  64
#define Mtot (Bb * Ss)   // 8192 tokens

typedef __attribute__((ext_vector_type(16))) __bf16 v16bf;
typedef __attribute__((ext_vector_type(8)))  float  v8f;
typedef __attribute__((ext_vector_type(4)))  unsigned int u32x4;

union FragBF { v16bf v; u32x4 q[2]; };

__device__ __forceinline__ unsigned short f2bf(float f) {
  unsigned u = __float_as_uint(f);
  u += 0x7FFFu + ((u >> 16) & 1u);        // round-to-nearest-even
  return (unsigned short)(u >> 16);
}

__device__ __forceinline__ v8f vzero8() {
  v8f v = {0.f, 0.f, 0.f, 0.f, 0.f, 0.f, 0.f, 0.f};
  return v;
}

// Per-lane WMMA fragment load (A operand 16x32, or B operand where data is
// stored N-major-row x K-contiguous, which is the same per-lane pattern).
// lanes 0-15 : row = lane,    K = 0..7 and 16..23
// lanes 16-31: row = lane-16, K = 8..15 and 24..31
__device__ __forceinline__ v16bf load_frag(const unsigned short* base, int ld, int lane) {
  int r  = lane & 15;
  int k0 = (lane >> 4) << 3;
  const unsigned short* p = base + r * ld + k0;
  FragBF f;
  f.q[0] = *(const u32x4*)(p);
  f.q[1] = *(const u32x4*)(p + 16);
  return f.v;
}

__device__ __forceinline__ v8f wmma_bf16(v16bf a, v16bf b, v8f c) {
  return __builtin_amdgcn_wmma_f32_16x16x32_bf16(false, a, false, b, (short)0, c, false, false);
}

// gfx1250 async copy: 16B global -> LDS, tracked by ASYNCcnt.
// (inline asm: portable across ROCm7.2 / amdgpu-toolchain builtin arity split)
__device__ __forceinline__ void async_cp16(unsigned short* lds, const unsigned short* g) {
  unsigned l32 = (unsigned)(unsigned long long)(uintptr_t)lds;   // LDS byte offset
  unsigned long long ga = (unsigned long long)(uintptr_t)g;
  asm volatile("global_load_async_to_lds_b128 %0, %1, off"
               :: "v"(l32), "v"(ga) : "memory");
}
__device__ __forceinline__ void wait_async0() { asm volatile("s_wait_asynccnt 0x0" ::: "memory"); }
__device__ __forceinline__ void wait_async2() { asm volatile("s_wait_asynccnt 0x2" ::: "memory"); }
__device__ __forceinline__ void wait_async4() { asm volatile("s_wait_asynccnt 0x4" ::: "memory"); }

// ---------------------------------------------------------------------------
// Kernel 1: generic f32 -> bf16 conversion (x and the 4 weight matrices, once)
// ---------------------------------------------------------------------------
__global__ __launch_bounds__(256) void cvt_bf16(const float* __restrict__ s,
                                                unsigned short* __restrict__ d, int n4) {
  int i = blockIdx.x * blockDim.x + threadIdx.x;       // float4 units
  if (i >= n4) return;
  float4 f = ((const float4*)s)[i];
  uint2 o;
  o.x = (unsigned)f2bf(f.x) | ((unsigned)f2bf(f.y) << 16);
  o.y = (unsigned)f2bf(f.z) | ((unsigned)f2bf(f.w) << 16);
  ((uint2*)d)[i] = o;
}

// ---------------------------------------------------------------------------
// Kernel 2: fused QKV projection.  Y = x @ W^T + b  (M=8192, N=1024, K=1024)
// blockIdx.z selects {Q, K, V}.  Q,K stored [B,NH,S,DH] bf16; V stored
// transposed [B,NH,DH,S] bf16 so the PV GEMM's B-operand is contiguous.
// Block tile 128x128, 8 waves (2x4), each wave 64x32 (4x2 WMMA tiles).
// A (bf16 x) double-buffered into LDS via async-to-LDS; B straight from W.
// ---------------------------------------------------------------------------
__global__ __launch_bounds__(256) void qkv_proj(
    const unsigned short* __restrict__ Xb,
    const unsigned short* __restrict__ wqb, const unsigned short* __restrict__ wkb,
    const unsigned short* __restrict__ wvb,
    const float* __restrict__ bq, const float* __restrict__ bk,
    const float* __restrict__ bvv,
    unsigned short* __restrict__ Q, unsigned short* __restrict__ K,
    unsigned short* __restrict__ Vt) {
  __shared__ unsigned short atile[2][128][40];   // 40-elem stride: 80B, conflict-free

  const int z = blockIdx.z;
  const unsigned short* W = (z == 0) ? wqb : (z == 1) ? wkb : wvb;
  const float* bias       = (z == 0) ? bq  : (z == 1) ? bk  : bvv;

  const int tid  = threadIdx.x;
  const int lane = tid & 31, wave = tid >> 5;
  const int wm = wave >> 2, wn = wave & 3;     // 2 x 4 wave grid
  const int m0 = blockIdx.y * 128;             // token base
  const int n0 = blockIdx.x * 128;             // output-feature base

  v8f acc[4][2];
#pragma unroll
  for (int i = 0; i < 4; i++)
#pragma unroll
    for (int j = 0; j < 2; j++) acc[i][j] = vzero8();

  // async A-tile staging: 128x32 bf16 = 512 16B-chunks, 2 per thread
  const int arow0 = tid >> 2,        arow1 = (tid + 256) >> 2;    // 0..63 / 64..127
  const int acc0  = (tid & 3) * 8,   acc1  = ((tid + 256) & 3) * 8;

#define ISSUE_A(kk_, b_)                                                        \
  do {                                                                          \
    async_cp16(&atile[b_][arow0][acc0], Xb + (m0 + arow0) * Hh + (kk_) + acc0); \
    async_cp16(&atile[b_][arow1][acc1], Xb + (m0 + arow1) * Hh + (kk_) + acc1); \
  } while (0)

  int buf = 0;
  ISSUE_A(0, 0);
  for (int kk = 0; kk < Hh; kk += 32) {
    if (kk + 32 < Hh) { ISSUE_A(kk + 32, buf ^ 1); wait_async2(); }
    else              { wait_async0(); }
    __syncthreads();                       // all waves' tile-kk chunks landed

    v16bf bfrag[2];
#pragma unroll
    for (int nt = 0; nt < 2; nt++)
      bfrag[nt] = load_frag(W + (n0 + wn * 32 + nt * 16) * Hh + kk, Hh, lane);
#pragma unroll
    for (int mt = 0; mt < 4; mt++) {
      v16bf afrag = load_frag(&atile[buf][wm * 64 + mt * 16][0], 40, lane);
#pragma unroll
      for (int nt = 0; nt < 2; nt++)
        acc[mt][nt] = wmma_bf16(afrag, bfrag[nt], acc[mt][nt]);
    }
    __syncthreads();                       // done reading before next overwrite
    buf ^= 1;
  }
#undef ISSUE_A

  // epilogue: bias add, bf16 convert, scatter to head-major layouts
#pragma unroll
  for (int nt = 0; nt < 2; nt++) {
    int gn   = n0 + wn * 32 + nt * 16 + (lane & 15);
    float bb = bias[gn];
    int head = gn >> 6, d = gn & 63;
#pragma unroll
    for (int mt = 0; mt < 4; mt++) {
#pragma unroll
      for (int e = 0; e < 8; e++) {
        int gm = m0 + wm * 64 + mt * 16 + e + ((lane >> 4) << 3);
        int bi = gm >> 9, si = gm & (Ss - 1);
        unsigned short val = f2bf(acc[mt][nt][e] + bb);
        if (z == 0)      Q [(((bi * NHh + head) * Ss + si) << 6) + d] = val;
        else if (z == 1) K [(((bi * NHh + head) * Ss + si) << 6) + d] = val;
        else             Vt[(((bi * NHh + head) * DHh + d) << 9) + si] = val;
      }
    }
  }
}

// ---------------------------------------------------------------------------
// Kernel 3: flash attention per (head, 128-query block).
// K and V^T 64x64 tiles are SHARED by all 8 waves -> cooperatively async-copied
// into double-buffered LDS (8x less global traffic than per-wave fetch).
// Each wave owns 16 query rows: QK^T (8 WMMA) -> masked online softmax (f32)
// -> P staged through wave-private LDS -> P@V (8 WMMA).
// Ragged cls_len: skip k-blocks past L (uniform per block); zero padded rows.
// ---------------------------------------------------------------------------
__global__ __launch_bounds__(256) void flash_attn(
    const unsigned short* __restrict__ Q, const unsigned short* __restrict__ K,
    const unsigned short* __restrict__ Vt, const int* __restrict__ cls_len,
    unsigned short* __restrict__ O) {
  __shared__ unsigned short ktile[2][64][72];   // 72-elem stride: 144B, conflict-free
  __shared__ unsigned short vtile[2][64][72];
  __shared__ unsigned short plds[8][16][72];    // per-wave P tile

  const int tid  = threadIdx.x;
  const int lane = tid & 31, wave = tid >> 5;
  const int hl = blockIdx.y;                 // 0..255 = b*NH + h
  const int bi = hl >> 4, hi = hl & 15;
  const int L  = cls_len[bi];
  const int q0b = blockIdx.x * 128;

  if (q0b >= L) {                            // whole block is padding: out = 0
    for (int it = tid; it < 128 * 8; it += 256) {
      int r = it >> 3, c = (it & 7) * 8;
      u32x4 zz = {0u, 0u, 0u, 0u};
      *(u32x4*)&O[(bi * Ss + q0b + r) * Hh + hi * DHh + c] = zz;
    }
    return;
  }

  const int q0 = q0b + wave * 16;
  const unsigned short* qptr = Q  + ((bi * NHh + hi) * Ss) * DHh;
  const unsigned short* kptr = K  + ((bi * NHh + hi) * Ss) * DHh;
  const unsigned short* vptr = Vt + ((bi * NHh + hi) * DHh) * Ss;

  // Q fragments for this wave's 16 rows (K-dim = d, two 32-wide steps)
  v16bf qf[2];
  qf[0] = load_frag(qptr + q0 * DHh + 0,  DHh, lane);
  qf[1] = load_frag(qptr + q0 * DHh + 32, DHh, lane);

  // cooperative K/V tile staging: each tile 64 rows x 8 chunks = 512 chunks
  const int tr0 = tid >> 3,        tr1 = (tid + 256) >> 3;   // 0..31 / 32..63
  const int tc0 = (tid & 7) * 8,   tc1 = ((tid + 256) & 7) * 8;

#define ISSUE_KV(kc_, b_)                                                        \
  do {                                                                           \
    async_cp16(&ktile[b_][tr0][tc0], kptr + ((kc_) + tr0) * DHh + tc0);          \
    async_cp16(&ktile[b_][tr1][tc1], kptr + ((kc_) + tr1) * DHh + tc1);          \
    async_cp16(&vtile[b_][tr0][tc0], vptr + tr0 * Ss + (kc_) + tc0);             \
    async_cp16(&vtile[b_][tr1][tc1], vptr + tr1 * Ss + (kc_) + tc1);             \
  } while (0)

  v8f o[4];
#pragma unroll
  for (int nt = 0; nt < 4; nt++) o[nt] = vzero8();
  float m[8], l[8];
#pragma unroll
  for (int e = 0; e < 8; e++) { m[e] = -3.0e38f; l[e] = 0.f; }
  const float scale = 0.125f;               // 1/sqrt(DH)

  const int jend = (L + 63) >> 6;           // >=1 here; uniform per block
  int buf = 0;
  ISSUE_KV(0, 0);
  for (int j = 0; j < jend; j++) {
    const int kc0 = j * 64;
    if (j + 1 < jend) { ISSUE_KV(kc0 + 64, buf ^ 1); wait_async4(); }
    else              { wait_async0(); }
    __syncthreads();                         // tile j visible to all waves

    // ---- scores: S = Q @ K^T for 16x64 tile -------------------------------
    v8f s[4];
#pragma unroll
    for (int nt = 0; nt < 4; nt++) {
      s[nt] = vzero8();
      v16bf kf0 = load_frag(&ktile[buf][nt * 16][0],  72, lane);
      v16bf kf1 = load_frag(&ktile[buf][nt * 16][32], 72, lane);
      s[nt] = wmma_bf16(qf[0], kf0, s[nt]);
      s[nt] = wmma_bf16(qf[1], kf1, s[nt]);
    }

    // ---- scale + ragged mask ---------------------------------------------
#pragma unroll
    for (int nt = 0; nt < 4; nt++) {
      int col = kc0 + nt * 16 + (lane & 15);
      bool ok = col < L;
#pragma unroll
      for (int e = 0; e < 8; e++)
        s[nt][e] = ok ? s[nt][e] * scale : -1.0e30f;
    }

    // ---- online softmax (row = accumulator element e, per 16-lane half) ---
#pragma unroll
    for (int e = 0; e < 8; e++) {
      float mx = fmaxf(fmaxf(s[0][e], s[1][e]), fmaxf(s[2][e], s[3][e]));
#pragma unroll
      for (int off = 1; off < 16; off <<= 1) mx = fmaxf(mx, __shfl_xor(mx, off, 32));
      float mnew  = fmaxf(m[e], mx);
      float alpha = __expf(m[e] - mnew);
      float rs = 0.f;
#pragma unroll
      for (int nt = 0; nt < 4; nt++) {
        float p = __expf(s[nt][e] - mnew);
        s[nt][e] = p;
        rs += p;
      }
#pragma unroll
      for (int off = 1; off < 16; off <<= 1) rs += __shfl_xor(rs, off, 32);
      l[e] = l[e] * alpha + rs;
      m[e] = mnew;
#pragma unroll
      for (int nt = 0; nt < 4; nt++) o[nt][e] *= alpha;
    }

    // ---- stage P (C-layout) -> wave-private LDS -> A-fragment layout ------
    // wave-local RAW: LDS ops of one wave execute in order; the asm fences
    // only stop the compiler from reordering typed LDS accesses.
    asm volatile("" ::: "memory");
#pragma unroll
    for (int nt = 0; nt < 4; nt++) {
      int col = nt * 16 + (lane & 15);
#pragma unroll
      for (int e = 0; e < 8; e++) {
        int row = e + ((lane >> 4) << 3);
        plds[wave][row][col] = f2bf(s[nt][e]);
      }
    }
    asm volatile("" ::: "memory");

    // ---- O += P @ V -------------------------------------------------------
#pragma unroll
    for (int step = 0; step < 2; step++) {
      v16bf pf = load_frag(&plds[wave][0][step * 32], 72, lane);
#pragma unroll
      for (int nt = 0; nt < 4; nt++) {
        v16bf vf = load_frag(&vtile[buf][nt * 16][step * 32], 72, lane);
        o[nt] = wmma_bf16(pf, vf, o[nt]);
      }
    }
    __syncthreads();                         // done with tile j before overwrite
    buf ^= 1;
  }
#undef ISSUE_KV

  // ---- normalize, zero padded rows, store bf16 to merged-head O -----------
#pragma unroll
  for (int nt = 0; nt < 4; nt++) {
    int d = nt * 16 + (lane & 15);
#pragma unroll
    for (int e = 0; e < 8; e++) {
      int row  = q0 + e + ((lane >> 4) << 3);
      float rl = (l[e] > 0.f) ? 1.0f / l[e] : 0.f;
      float v  = (row < L) ? o[nt][e] * rl : 0.f;
      O[(bi * Ss + row) * Hh + hi * DHh + d] = f2bf(v);
    }
  }
}

// ---------------------------------------------------------------------------
// Kernel 4: output projection  out = O @ wf^T + bf   (f32 result)
// All operands bf16 in global (L2-resident); per-lane b128 fragment loads.
// ---------------------------------------------------------------------------
__global__ __launch_bounds__(256) void out_proj(
    const unsigned short* __restrict__ A,    // [8192][1024] bf16
    const unsigned short* __restrict__ Wf,   // [1024][1024] bf16
    const float* __restrict__ bias, float* __restrict__ out) {
  const int tid  = threadIdx.x;
  const int lane = tid & 31, wave = tid >> 5;
  const int wm = wave >> 2, wn = wave & 3;
  const int m0 = blockIdx.y * 128, n0 = blockIdx.x * 128;

  v8f acc[4][2];
#pragma unroll
  for (int i = 0; i < 4; i++)
#pragma unroll
    for (int j = 0; j < 2; j++) acc[i][j] = vzero8();

  for (int kk = 0; kk < Hh; kk += 32) {
    v16bf bfrag[2];
#pragma unroll
    for (int nt = 0; nt < 2; nt++)
      bfrag[nt] = load_frag(Wf + (n0 + wn * 32 + nt * 16) * Hh + kk, Hh, lane);
#pragma unroll
    for (int mt = 0; mt < 4; mt++) {
      v16bf afrag = load_frag(A + (m0 + wm * 64 + mt * 16) * Hh + kk, Hh, lane);
#pragma unroll
      for (int nt = 0; nt < 2; nt++)
        acc[mt][nt] = wmma_bf16(afrag, bfrag[nt], acc[mt][nt]);
    }
  }

#pragma unroll
  for (int nt = 0; nt < 2; nt++) {
    int gn   = n0 + wn * 32 + nt * 16 + (lane & 15);
    float bb = bias[gn];
#pragma unroll
    for (int mt = 0; mt < 4; mt++) {
#pragma unroll
      for (int e = 0; e < 8; e++) {
        int gm = m0 + wm * 64 + mt * 16 + e + ((lane >> 4) << 3);
        out[gm * Hh + gn] = acc[mt][nt][e] + bb;
      }
    }
  }
}

// ---------------------------------------------------------------------------
extern "C" void kernel_launch(void* const* d_in, const int* in_sizes, int n_in,
                              void* d_out, int out_size, void* d_ws, size_t ws_size,
                              hipStream_t stream) {
  const float* x   = (const float*)d_in[0];
  const int*   cl  = (const int*)  d_in[1];
  const float* wq  = (const float*)d_in[2];
  const float* bq  = (const float*)d_in[3];
  const float* wk  = (const float*)d_in[4];
  const float* bk  = (const float*)d_in[5];
  const float* wv  = (const float*)d_in[6];
  const float* bv  = (const float*)d_in[7];
  const float* wf  = (const float*)d_in[8];
  const float* bf  = (const float*)d_in[9];

  // Workspace layout (88 MB total):
  //   [0,8MB)   : bf16 weights wq,wk,wv,wf (2MB each)
  //   [8,24MB)  : x   bf16 [B*S, H]
  //   [24,40MB) : Q   bf16 [B,NH,S,DH]
  //   [40,56MB) : K   bf16 [B,NH,S,DH]
  //   [56,72MB) : V^T bf16 [B,NH,DH,S]
  //   [72,88MB) : attn output bf16 [B,S,H]
  char* ws = (char*)d_ws;
  const size_t MB = 1u << 20;
  unsigned short* wqb = (unsigned short*)(ws + 0 * MB);
  unsigned short* wkb = (unsigned short*)(ws + 2 * MB);
  unsigned short* wvb = (unsigned short*)(ws + 4 * MB);
  unsigned short* wfb = (unsigned short*)(ws + 6 * MB);
  unsigned short* Xb  = (unsigned short*)(ws + 8 * MB);
  unsigned short* Qb  = (unsigned short*)(ws + 24 * MB);
  unsigned short* Kb  = (unsigned short*)(ws + 40 * MB);
  unsigned short* Vtb = (unsigned short*)(ws + 56 * MB);
  unsigned short* Ob  = (unsigned short*)(ws + 72 * MB);

  const int wn4 = Hh * Hh / 4, xn4 = Mtot * Hh / 4;
  cvt_bf16<<<(xn4 + 255) / 256, 256, 0, stream>>>(x,  Xb,  xn4);
  cvt_bf16<<<(wn4 + 255) / 256, 256, 0, stream>>>(wq, wqb, wn4);
  cvt_bf16<<<(wn4 + 255) / 256, 256, 0, stream>>>(wk, wkb, wn4);
  cvt_bf16<<<(wn4 + 255) / 256, 256, 0, stream>>>(wv, wvb, wn4);
  cvt_bf16<<<(wn4 + 255) / 256, 256, 0, stream>>>(wf, wfb, wn4);

  dim3 g2(Hh / 128, Mtot / 128, 3);           // 8 x 64 x 3
  qkv_proj<<<g2, 256, 0, stream>>>(Xb, wqb, wkb, wvb, bq, bk, bv, Qb, Kb, Vtb);

  dim3 g3(Ss / 128, Bb * NHh);                // 4 x 256
  flash_attn<<<g3, 256, 0, stream>>>(Qb, Kb, Vtb, cl, Ob);

  dim3 g4(Hh / 128, Mtot / 128);              // 8 x 64
  out_proj<<<g4, 256, 0, stream>>>(Ob, wfb, bf, (float*)d_out);
}
